// Injector_34248069219091
// MI455X (gfx1250) — matrix operand order repaired and checked by reading, not compiled, not run on hardware
//
#include <hip/hip_runtime.h>

#define DIMX   768
#define HEADS  6
#define CCH    128
#define LEVELS 3
#define POINTS 4
#define NBATCH 8
#define LQ     1024
#define LIN    5376
#define NCAT   224   // 144 offsets + 72 attn logits + 8 zero pad (multiple of 16)
#define KCHUNK 256   // K-chunk staged into LDS via TDM (DIMX % KCHUNK == 0)

typedef __attribute__((ext_vector_type(16))) __bf16         bf16x16;
typedef __attribute__((ext_vector_type(16))) unsigned short u16x16;
typedef __attribute__((ext_vector_type(8)))  unsigned short u16x8;
typedef __attribute__((ext_vector_type(8)))  float          f32x8;
typedef __attribute__((ext_vector_type(4)))  float          f32x4;
typedef __attribute__((ext_vector_type(4)))  unsigned int   u32x4;
typedef __attribute__((ext_vector_type(8)))  int            i32x8;
typedef __attribute__((ext_vector_type(4)))  int            i32x4;

union BFrag { u16x16 u; bf16x16 b; };

static __device__ inline unsigned short f2bf(float f) {
  unsigned u = __builtin_bit_cast(unsigned, f);
  u += 0x7FFFu + ((u >> 16) & 1u);          // round-to-nearest-even
  return (unsigned short)(u >> 16);
}

static __device__ inline f32x8 wmma_bf16(const BFrag& a, const BFrag& b, f32x8 c) {
  // v_wmma_f32_16x16x32_bf16 : D = A(16x32) * B(32x16) + C
  return __builtin_amdgcn_wmma_f32_16x16x32_bf16(
      false, a.b, false, b.b, (short)0, c, false, false);
}

// ---- Tensor Data Mover: 2D tile (global -> LDS), data_size = 8 bytes -----
// tile_x/stride in 8-byte units; rows packed contiguously into LDS.
// Must be called wave-uniform; EXEC is ignored by TDM.
static __device__ inline void tdm_load_2d(unsigned lds_byte_addr, const void* gptr,
                                          unsigned tile_x, unsigned tile_y,
                                          unsigned stride_x) {
  unsigned long long ga = (unsigned long long)(uintptr_t)gptr;
  u32x4 g0;
  g0[0] = 1u;                                              // count=1, user D#
  g0[1] = lds_byte_addr;                                   // lds_addr[31:0]
  g0[2] = (unsigned)ga;                                    // global_addr[31:0]
  g0[3] = (unsigned)((ga >> 32) & 0x01FFFFFFu) | (2u << 30); // addr[56:32]|type=2
  i32x8 g1;
  g1[0] = (int)(3u << 16);            // workgroup_mask=0, data_size=3 (8B)
  g1[1] = (int)(tile_x << 16);        // tensor_dim0[15:0]  (== tile_x, no OOB)
  g1[2] = (int)(tile_y << 16);        // tensor_dim0 hi=0 | tensor_dim1[15:0]
  g1[3] = (int)(tile_x << 16);        // tensor_dim1 hi=0 | tile_dim0
  g1[4] = (int)tile_y;                // tile_dim1 | tile_dim2=0
  g1[5] = (int)stride_x;              // tensor_dim0_stride[31:0]
  g1[6] = 0;                          // stride0 hi | dim1_stride lo (unused 2D)
  g1[7] = 0;
  i32x4 z4 = {0, 0, 0, 0};
  i32x8 z8 = {0, 0, 0, 0, 0, 0, 0, 0};
  __builtin_amdgcn_tensor_load_to_lds(g0, g1, z4, z4, z8, 0);
}

// ---------------------------------------------------------------- LayerNorm
// One wave32 per row of DIMX=768; each lane owns 24 strided elements.
__global__ __launch_bounds__(256) void k_ln_cast(
    const float* __restrict__ X, const float* __restrict__ w,
    const float* __restrict__ b, unsigned short* __restrict__ Y, int rows) {
  int wid  = (blockIdx.x * blockDim.x + threadIdx.x) >> 5;
  int lane = threadIdx.x & 31;
  if (wid >= rows) return;
  const float* x = X + (size_t)wid * DIMX;
  float v[24], s = 0.f, ss = 0.f;
#pragma unroll
  for (int i = 0; i < 24; ++i) {
    v[i] = x[lane + 32 * i];
    s += v[i]; ss += v[i] * v[i];
  }
#pragma unroll
  for (int off = 16; off > 0; off >>= 1) {
    s  += __shfl_xor(s,  off, 32);
    ss += __shfl_xor(ss, off, 32);
  }
  float mean = s * (1.f / DIMX);
  float var  = ss * (1.f / DIMX) - mean * mean;
  float rstd = rsqrtf(var + 1e-6f);
  unsigned short* y = Y + (size_t)wid * DIMX;
#pragma unroll
  for (int i = 0; i < 24; ++i) {
    int d = lane + 32 * i;
    y[d] = f2bf((v[i] - mean) * rstd * w[d] + b[d]);
  }
}

// ------------------------------------------------- weight transpose + cast
__global__ __launch_bounds__(256) void k_transpose_cast(
    const float* __restrict__ W, unsigned short* __restrict__ Wt, int K, int N) {
  int idx = blockIdx.x * blockDim.x + threadIdx.x;
  if (idx >= K * N) return;
  int n = idx / K, k = idx - n * K;
  Wt[(size_t)n * K + k] = f2bf(W[(size_t)k * N + n]);
}

__global__ __launch_bounds__(256) void k_build_cat(
    const float* __restrict__ Wo, const float* __restrict__ bo,
    const float* __restrict__ Wa, const float* __restrict__ ba,
    unsigned short* __restrict__ Wt, float* __restrict__ bcat) {
  int idx = blockIdx.x * blockDim.x + threadIdx.x;
  if (idx >= NCAT * DIMX) return;
  int n = idx / DIMX, k = idx - n * DIMX;
  float v = 0.f;
  if (n < 144)      v = Wo[(size_t)k * 144 + n];
  else if (n < 216) v = Wa[(size_t)k * 72 + (n - 144)];
  Wt[(size_t)n * DIMX + k] = f2bf(v);
  if (k == 0) {
    float bb = 0.f;
    if (n < 144)      bb = bo[n];
    else if (n < 216) bb = ba[n - 144];
    bcat[n] = bb;
  }
}

// --------------------------------------------------------- WMMA GEMM 16x64
// C(MxDIMX) = A(MxDIMX bf16 rowmajor) * Bt(DIMXxDIMX bf16 == B^T) + bias
// 256 threads = 8 waves; wave computes 16(M) x 64(N); K fixed = DIMX.
// B panel (64 cols x KCHUNK) is TDM-staged into double-buffered LDS and the
// DMA for chunk c+1 overlaps the WMMAs of chunk c (TENSORcnt double buffer).
// fuse==1: C = resid + gamma * (acc + bias)
__global__ __launch_bounds__(256) void k_gemm16x64(
    const unsigned short* __restrict__ A, const unsigned short* __restrict__ Bt,
    const float* __restrict__ bias, float* __restrict__ C,
    int M, int N, int fuse,
    const float* __restrict__ resid, const float* __restrict__ gamma) {
  __shared__ unsigned short ldsB[2][64 * KCHUNK];   // 2 x 32 KB

  const int lane = threadIdx.x & 31;
  const int wave = threadIdx.x >> 5;
  const int m0   = (blockIdx.x * 8 + wave) * 16;
  const int n0   = blockIdx.y * 64;
  const int half = lane >> 4;
  const int l16  = lane & 15;
  const int K    = DIMX;
  const int NCH  = DIMX / KCHUNK;                   // 3 chunks

  // A fragment (16-bit A 16x32): lane-half kbase = half*16,
  // elems 0..15 = K[kb..kb+15]
  const size_t arow = (size_t)(m0 + l16) * K + half * 16;

  f32x8 acc[4] = {{}, {}, {}, {}};

  // prologue: DMA chunk 0 -> buffer 0
  if (threadIdx.x < 32) {
    tdm_load_2d((unsigned)(uintptr_t)&ldsB[0][0],
                Bt + (size_t)n0 * K, /*tile_x=*/KCHUNK * 2 / 8, /*tile_y=*/64,
                /*stride_x=*/K * 2 / 8);
  }

  for (int c = 0; c < NCH; ++c) {
    if (threadIdx.x < 32) {
      if (c + 1 < NCH) {
        tdm_load_2d((unsigned)(uintptr_t)&ldsB[(c + 1) & 1][0],
                    Bt + (size_t)n0 * K + (c + 1) * KCHUNK,
                    KCHUNK * 2 / 8, 64, K * 2 / 8);
        __builtin_amdgcn_s_wait_tensorcnt(1);   // chunk c landed; c+1 in flight
      } else {
        __builtin_amdgcn_s_wait_tensorcnt(0);
      }
    }
    __syncthreads();                            // chunk c visible to all waves

    const unsigned short* lb = &ldsB[c & 1][0];
    const int kg = c * KCHUNK;
#pragma unroll
    for (int kk = 0; kk < KCHUNK; kk += 32) {
      BFrag a;
      u16x8 alo = *(const u16x8*)(A + arow + kg + kk);
      u16x8 ahi = *(const u16x8*)(A + arow + kg + kk + 8);
      a.u = __builtin_shufflevector(alo, ahi,
          0,1,2,3,4,5,6,7,8,9,10,11,12,13,14,15);
      __builtin_prefetch(A + arow + kg + kk + 128, 0, 1);
#pragma unroll
      for (int j = 0; j < 4; ++j) {
        // B frag: column (local) = j*16 + l16; lanes 0-15 K 0..15, 16-31 K 16..31
        const unsigned short* bp = lb + (j * 16 + l16) * KCHUNK + kk + half * 16;
        BFrag bf;
        u16x8 blo = *(const u16x8*)(bp);
        u16x8 bhi = *(const u16x8*)(bp + 8);
        bf.u = __builtin_shufflevector(blo, bhi,
            0,1,2,3,4,5,6,7,8,9,10,11,12,13,14,15);
        acc[j] = wmma_bf16(a, bf, acc[j]);
      }
    }
    __syncthreads();                            // protect buffer reuse
  }

  // C/D layout: lane 0-15 -> N=lane, VGPR r -> M=r (+8 for lanes 16-31)
#pragma unroll
  for (int j = 0; j < 4; ++j) {
    int nn = n0 + j * 16 + l16;
    float bb = bias[nn];
#pragma unroll
    for (int r = 0; r < 8; ++r) {
      int m = m0 + r + half * 8;
      size_t o = (size_t)m * N + nn;
      float v = acc[j][r] + bb;
      if (fuse) v = resid[o] + gamma[nn] * v;
      C[o] = v;
    }
  }
}

// ------------------------------------------------- WMMA GEMM 16x16 per wave
__global__ __launch_bounds__(256) void k_gemm16x16(
    const unsigned short* __restrict__ A, const unsigned short* __restrict__ Bt,
    const float* __restrict__ bias, float* __restrict__ C,
    int M, int N, int K) {
  const int lane = threadIdx.x & 31;
  const int wave = threadIdx.x >> 5;
  const int m0   = (blockIdx.x * 8 + wave) * 16;
  const int n0   = blockIdx.y * 16;
  const int half = lane >> 4;
  const int l16  = lane & 15;
  const size_t arow = (size_t)(m0 + l16) * K + half * 16;
  const unsigned short* bp0 = Bt + (size_t)(n0 + l16) * K + half * 16;
  f32x8 acc = {};
  for (int kk = 0; kk < K; kk += 32) {
    BFrag a, bf;
    u16x8 alo = *(const u16x8*)(A + arow + kk);
    u16x8 ahi = *(const u16x8*)(A + arow + kk + 8);
    a.u = __builtin_shufflevector(alo, ahi,
        0,1,2,3,4,5,6,7,8,9,10,11,12,13,14,15);
    u16x8 blo = *(const u16x8*)(bp0 + kk);
    u16x8 bhi = *(const u16x8*)(bp0 + kk + 8);
    bf.u = __builtin_shufflevector(blo, bhi,
        0,1,2,3,4,5,6,7,8,9,10,11,12,13,14,15);
    acc = wmma_bf16(a, bf, acc);
  }
#pragma unroll
  for (int r = 0; r < 8; ++r) {
    int m = m0 + r + half * 8;
    int nn = n0 + l16;
    C[(size_t)m * N + nn] = acc[r] + bias[nn];
  }
}

// ----------------------------------------------------- deformable sampling
// one wave per (n, q, head); lane owns 4 of 128 channels (float4 gathers)
__global__ __launch_bounds__(256) void k_sample(
    const float* __restrict__ value, const float* __restrict__ oa_all,
    const float* __restrict__ refp, unsigned short* __restrict__ outbf) {
  const int wid  = (blockIdx.x * blockDim.x + threadIdx.x) >> 5;
  const int lane = threadIdx.x & 31;
  const int h    = wid % HEADS;
  const int row  = wid / HEADS;            // n*LQ + q
  const int n    = row / LQ;
  const float* oa = oa_all + (size_t)row * NCAT;

  float w[12];
  float mx = -1e30f;
#pragma unroll
  for (int i = 0; i < 12; ++i) { w[i] = oa[144 + h * 12 + i]; mx = fmaxf(mx, w[i]); }
  float sum = 0.f;
#pragma unroll
  for (int i = 0; i < 12; ++i) { w[i] = __expf(w[i] - mx); sum += w[i]; }
  const float inv = 1.f / sum;

  const int HH[3] = {64, 32, 16};
  const int WW[3] = {64, 32, 16};
  const int SS[3] = {0, 4096, 5120};
  f32x4 acc = {};
  const float* vbase = value + (size_t)n * LIN * DIMX + h * CCH + lane * 4;

#pragma unroll
  for (int l = 0; l < LEVELS; ++l) {
    const int   Hs = HH[l], Ws = WW[l], st = SS[l];
    const float fW = (float)Ws, fH = (float)Hs;
    const float rx = refp[((size_t)row * LEVELS + l) * 2 + 0];
    const float ry = refp[((size_t)row * LEVELS + l) * 2 + 1];
#pragma unroll
    for (int p = 0; p < POINTS; ++p) {
      int oc = ((h * LEVELS + l) * POINTS + p) * 2;
      float x = (rx + oa[oc]     / fW) * fW - 0.5f;
      float y = (ry + oa[oc + 1] / fH) * fH - 0.5f;
      float x0f = floorf(x), y0f = floorf(y);
      float dx = x - x0f, dy = y - y0f;
      int x0 = (int)x0f, y0 = (int)y0f;
      float aw = w[l * POINTS + p] * inv;
      float cw[4] = {(1.f - dx) * (1.f - dy), dx * (1.f - dy),
                     (1.f - dx) * dy,         dx * dy};
      int cx[4] = {x0, x0 + 1, x0,     x0 + 1};
      int cy[4] = {y0, y0,     y0 + 1, y0 + 1};
#pragma unroll
      for (int c = 0; c < 4; ++c) {
        int xi = cx[c], yi = cy[c];
        bool ok = (xi >= 0) & (xi < Ws) & (yi >= 0) & (yi < Hs);
        float wgt = ok ? cw[c] * aw : 0.f;
        int xc = min(max(xi, 0), Ws - 1);
        int yc = min(max(yi, 0), Hs - 1);
        f32x4 gv = *(const f32x4*)(vbase + (size_t)(st + yc * Ws + xc) * DIMX);
        acc += gv * wgt;
      }
    }
  }
  unsigned short* o = outbf + (size_t)row * DIMX + h * CCH + lane * 4;
  o[0] = f2bf(acc[0]); o[1] = f2bf(acc[1]);
  o[2] = f2bf(acc[2]); o[3] = f2bf(acc[3]);
}

// ---------------------------------------------------------------- launcher
extern "C" void kernel_launch(void* const* d_in, const int* in_sizes, int n_in,
                              void* d_out, int out_size, void* d_ws, size_t ws_size,
                              hipStream_t stream) {
  const float* query = (const float*)d_in[0];
  const float* refp  = (const float*)d_in[1];
  const float* feat  = (const float*)d_in[2];
  const float* qn_w  = (const float*)d_in[5];
  const float* qn_b  = (const float*)d_in[6];
  const float* fn_w  = (const float*)d_in[7];
  const float* fn_b  = (const float*)d_in[8];
  const float* Wv    = (const float*)d_in[9];
  const float* bv    = (const float*)d_in[10];
  const float* Wo    = (const float*)d_in[11];
  const float* bo    = (const float*)d_in[12];
  const float* Wa    = (const float*)d_in[13];
  const float* ba    = (const float*)d_in[14];
  const float* Wout  = (const float*)d_in[15];
  const float* bout  = (const float*)d_in[16];
  const float* gamma = (const float*)d_in[17];
  float* out = (float*)d_out;

  const int rowsF = NBATCH * LIN;   // 43008
  const int rowsQ = NBATCH * LQ;    // 8192

  char* ws = (char*)d_ws;
  size_t off = 0;
  auto take = [&](size_t bytes) {
    char* p = ws + off;
    off += (bytes + 255) & ~(size_t)255;
    return p;
  };
  unsigned short* f_bf  = (unsigned short*)take((size_t)rowsF * DIMX * 2);
  unsigned short* q_bf  = (unsigned short*)take((size_t)rowsQ * DIMX * 2);
  float*          value = (float*)take((size_t)rowsF * DIMX * 4);
  unsigned short* Wv_t  = (unsigned short*)take((size_t)DIMX * DIMX * 2);
  unsigned short* Wo_t  = (unsigned short*)take((size_t)DIMX * DIMX * 2);
  unsigned short* Wc_t  = (unsigned short*)take((size_t)NCAT * DIMX * 2);
  float*          bcat  = (float*)take((size_t)NCAT * 4);
  float*          offaw = (float*)take((size_t)rowsQ * NCAT * 4);
  unsigned short* o_bf  = (unsigned short*)take((size_t)rowsQ * DIMX * 2);
  (void)ws_size;

  // 1) LayerNorms -> bf16
  k_ln_cast<<<rowsF / 8, 256, 0, stream>>>(feat, fn_w, fn_b, f_bf, rowsF);
  k_ln_cast<<<rowsQ / 8, 256, 0, stream>>>(query, qn_w, qn_b, q_bf, rowsQ);

  // 2) weight prep (transpose + bf16 cast)
  k_transpose_cast<<<(DIMX * DIMX + 255) / 256, 256, 0, stream>>>(Wv, Wv_t, DIMX, DIMX);
  k_transpose_cast<<<(DIMX * DIMX + 255) / 256, 256, 0, stream>>>(Wout, Wo_t, DIMX, DIMX);
  k_build_cat<<<(NCAT * DIMX + 255) / 256, 256, 0, stream>>>(Wo, bo, Wa, ba, Wc_t, bcat);

  // 3) value = LN(feat) @ Wv + bv   (M=43008, N=768, K=768)
  k_gemm16x64<<<dim3(rowsF / 128, DIMX / 64), 256, 0, stream>>>(
      f_bf, Wv_t, bv, value, rowsF, DIMX, 0, nullptr, nullptr);

  // 4) [offs|aw] = LN(q) @ [Wo|Wa]  (M=8192, N=224, K=768)
  k_gemm16x16<<<dim3(rowsQ / 128, NCAT / 16), 256, 0, stream>>>(
      q_bf, Wc_t, bcat, offaw, rowsQ, NCAT, DIMX);

  // 5) deformable bilinear sampling -> bf16 heads output
  k_sample<<<(rowsQ * HEADS) / 8, 256, 0, stream>>>(value, offaw, refp, o_bf);

  // 6) out = query + gamma * (heads @ Wout + bout)
  k_gemm16x64<<<dim3(rowsQ / 128, DIMX / 64), 256, 0, stream>>>(
      o_bf, Wo_t, bout, out, rowsQ, DIMX, 1, query, gamma);
}
